// MapNet_60430189854988
// MI455X (gfx1250) — matrix operand order repaired and checked by reading, not compiled, not run on hardware
//
#include <hip/hip_runtime.h>
#include <hip/hip_bf16.h>
#include <math.h>

// ---------------- problem constants ----------------
#define EE    128           // hidden / channel dim
#define MSZ   41            // map size
#define MM    1681          // 41*41
#define BB    16            // batch
#define TT    16            // image time steps
#define NSTEP 15            // scan steps = T-1
#define OO    4             // orientations
#define KW    15            // kernel width
#define KK    225           // 15*15
#define VR    7             // view range / pad
#define LOC   27            // 41-15+1 localize output
#define NX    (BB*MM)       // 26896 LSTM rows (= 16*1681, divisible by 16)
#define NPAD  55            // 41 + 2*7

typedef __attribute__((ext_vector_type(16))) __bf16 v16bf;
typedef __attribute__((ext_vector_type(8)))  __bf16 v8bf;
typedef __attribute__((ext_vector_type(8)))  float  v8f;
typedef __attribute__((ext_vector_type(4)))  float  f4;

__device__ __forceinline__ float sigm(float x){ return 1.0f/(1.0f+expf(-x)); }

// ---------------- pose init: delta at center, o=0 ----------------
__global__ void k_init_pose(float* pose){
  int b = threadIdx.x;
  if (b < BB) pose[((b*OO + 0)*MSZ + 20)*MSZ + 20] = 1.0f;
}

// ---------------- weights: bf16 [col][256] = [W_ih | W_hh] ----------------
__global__ void k_wcat(const float* __restrict__ wih, const float* __restrict__ whh,
                       __bf16* __restrict__ Wcat){
  int g = blockIdx.x*256 + threadIdx.x;           // 512*256 threads
  int k = g & 255, c = g >> 8;
  float v = (k < EE) ? wih[c*EE + k] : whh[c*EE + (k - EE)];
  Wcat[g] = (__bf16)v;
}

// ---------------- per-step conv weights, rotation/flip resolved ----------------
// wreg[b][o][u][v][e] = rot_o(img_s[e])[14-u,14-v]   (register: convT folded to conv)
// wloc[b][o][u][v][e] = rot_o(img_{s+1}[e])[u,v]     (localize: direct conv)
__global__ void k_prep_w(const float* __restrict__ images,
                         float* __restrict__ wreg, float* __restrict__ wloc, int s){
  int g = blockIdx.x*256 + threadIdx.x;
  if (g >= 2*BB*OO*KK*EE) return;
  int e    = g & (EE-1);
  int rest = g >> 7;
  int uv   = rest % KK; rest /= KK;
  int o    = rest & 3;  rest >>= 2;
  int b    = rest & 15; rest >>= 4;
  int which = rest;                       // 0 -> wreg (t=s), 1 -> wloc (t=s+1)
  int u = uv / KW, v = uv % KW;
  int i1, i2;
  if (which == 0){
    switch(o){ case 0: i1=14-u; i2=14-v; break;
               case 1: i1=14-v; i2=u;    break;
               case 2: i1=u;    i2=v;    break;
               default: i1=v;   i2=14-u; break; }
  } else {
    switch(o){ case 0: i1=u;    i2=v;    break;
               case 1: i1=v;    i2=14-u; break;
               case 2: i1=14-u; i2=14-v; break;
               default: i1=14-v; i2=u;   break; }
  }
  int t = s + which;
  float val = images[(((size_t)(b*TT + t)*EE + e)*KW + i1)*KW + i2];
  float* dst = which ? wloc : wreg;
  dst[(((size_t)(b*OO + o)*KK) + uv)*EE + e] = val;
}

// ---------------- register conv: pose [4,41,41] -> X rows [(b,y,x)][e] ----------------
__global__ void __launch_bounds__(256)
k_register(const float* __restrict__ pose, const float* __restrict__ wreg,
           float* __restrict__ X){
  __shared__ float sp[OO][NPAD][NPAD];            // 48.4 KB padded pose for this b
  const int b = blockIdx.y, tid = threadIdx.x;
  for (int idx = tid; idx < OO*NPAD*NPAD; idx += 256){
    int o = idx / (NPAD*NPAD); int r = idx - o*NPAD*NPAD;
    int yy = r / NPAD, xx = r % NPAD;
    int y = yy - VR, x = xx - VR;
    float v = 0.f;
    if ((unsigned)y < MSZ && (unsigned)x < MSZ)
      v = pose[((b*OO + o)*MSZ + y)*MSZ + x];
    sp[o][yy][xx] = v;
  }
  __syncthreads();
  int pix = blockIdx.x*16 + (tid >> 4);
  int ez  = tid & 15;                              // 8-channel chunk
  if (pix >= MM) return;
  int y = pix / MSZ, x = pix % MSZ;
  f4 acc0 = {0.f,0.f,0.f,0.f}, acc1 = {0.f,0.f,0.f,0.f};
  const float* wb = wreg + (size_t)b*OO*KK*EE + ez*8;
  for (int o = 0; o < OO; ++o)
    for (int u = 0; u < KW; ++u){
      const float* prow = &sp[o][y+u][x];
      const float* wrow = wb + (size_t)(o*KK + u*KW)*EE;
      #pragma unroll
      for (int v = 0; v < KW; ++v){
        float p = prow[v];
        const f4* wp = reinterpret_cast<const f4*>(wrow + (size_t)v*EE);
        acc0 += p * wp[0];
        acc1 += p * wp[1];
      }
    }
  float* xr = X + ((size_t)(b*MM + pix))*EE + ez*8;
  *reinterpret_cast<f4*>(xr)     = acc0;
  *reinterpret_cast<f4*>(xr + 4) = acc1;
}

// ---------------- pack A = bf16([X | H]) rows of 256 ----------------
__global__ void k_pack(const float* __restrict__ X, const float* __restrict__ H,
                       __bf16* __restrict__ Abuf){
  size_t g = (size_t)blockIdx.x*256 + threadIdx.x;
  if (g >= (size_t)NX*256) return;
  int k = (int)(g & 255);
  size_t row = g >> 8;
  float v = (k < EE) ? X[row*EE + k] : H[row*EE + (k - EE)];
  Abuf[g] = (__bf16)v;
}

// ---------------- WMMA GEMM + fused LSTM pointwise ----------------
// Block = one 16-row tile (grid = 1681). Wave w handles e-tile w: columns
// {w*16+e, +128, +256, +384} -> all four gates land in matching lane/VGPR slots.
__global__ void __launch_bounds__(256)
k_gemm_lstm(const __bf16* __restrict__ Abuf, const __bf16* __restrict__ Wcat,
            const float* __restrict__ bih, const float* __restrict__ bhh,
            float* __restrict__ H, float* __restrict__ C,
            float* __restrict__ maps, int s){
  const int lane = threadIdx.x & 31;
  const int et   = threadIdx.x >> 5;         // e-tile 0..7
  const int row0 = blockIdx.x * 16;
  const int m    = lane & 15;
  const int kh   = lane >> 4;                // K-half select per ISA layout
  const int e    = et*16 + m;                // output channel (also B-matrix column)

  v8f acc[4];
  const v8f zer = {0.f,0.f,0.f,0.f,0.f,0.f,0.f,0.f};
  #pragma unroll
  for (int g = 0; g < 4; ++g) acc[g] = zer;

  const __bf16* arow = Abuf + (size_t)(row0 + m)*256;
  #pragma unroll
  for (int kb = 0; kb < 256; kb += 32){
    if (kb < 224) __builtin_prefetch(arow + kb + 32, 0, 0);  // gfx1250 global_prefetch
    // A 16x32 bf16 fragment: two contiguous 8-elem runs per lane (ISA 7.12.2)
    v8bf alo = *reinterpret_cast<const v8bf*>(arow + kb + kh*8);
    v8bf ahi = *reinterpret_cast<const v8bf*>(arow + kb + 16 + kh*8);
    v16bf a;
    #pragma unroll
    for (int j = 0; j < 8; ++j){ a[j] = alo[j]; a[8+j] = ahi[j]; }
    #pragma unroll
    for (int g = 0; g < 4; ++g){
      // B 32x16 fragment: lane holds column e, two contiguous 8-elem K runs (same as A)
      const __bf16* wrow = Wcat + (size_t)(g*EE + e)*256;
      v8bf blo = *reinterpret_cast<const v8bf*>(wrow + kb + kh*8);
      v8bf bhi = *reinterpret_cast<const v8bf*>(wrow + kb + 16 + kh*8);
      v16bf bv;
      #pragma unroll
      for (int j = 0; j < 8; ++j){ bv[j] = blo[j]; bv[8+j] = bhi[j]; }
      acc[g] = __builtin_amdgcn_wmma_f32_16x16x32_bf16(
                 false, a, false, bv, (short)0, acc[g], false, false);
    }
  }

  float bg[4];
  #pragma unroll
  for (int g = 0; g < 4; ++g) bg[g] = bih[g*EE + e] + bhh[g*EE + e];

  #pragma unroll
  for (int r = 0; r < 8; ++r){               // C/D tile: row = row0 + kh*8 + r, col = e
    int row = row0 + kh*8 + r;
    float gi = acc[0][r] + bg[0], gf = acc[1][r] + bg[1];
    float gg = acc[2][r] + bg[2], go = acc[3][r] + bg[3];
    size_t ix = (size_t)row*EE + e;
    float c2 = sigm(gf)*C[ix] + sigm(gi)*tanhf(gg);
    float h2 = sigm(go)*tanhf(c2);
    C[ix] = c2; H[ix] = h2;
    int bb = row / MM, pix = row - bb*MM;
    maps[(((size_t)(bb*NSTEP + s))*EE + e)*MM + pix] = h2;
  }
}

// ---------------- localize conv: wave per (b,o,yy,xx), lanes over e ----------------
__global__ void __launch_bounds__(256)
k_localize(const float* __restrict__ H, const float* __restrict__ wloc,
           float* __restrict__ raw_ws, float* __restrict__ raws_out, int s){
  int gid  = blockIdx.x*8 + (threadIdx.x >> 5);
  int lane = threadIdx.x & 31;
  if (gid >= BB*OO*LOC*LOC) return;
  int xx = gid % LOC; int t1 = gid / LOC;
  int yy = t1 % LOC;  t1 /= LOC;
  int o  = t1 & 3;    int b  = t1 >> 2;
  float acc = 0.f;
  const float* wbase = wloc + (size_t)(b*OO + o)*KK*EE;
  for (int u = 0; u < KW; ++u){
    const float* mrow = H + (size_t)(b*MM + (yy+u)*MSZ + xx)*EE;
    const float* wrow = wbase + (size_t)(u*KW)*EE;
    for (int v = 0; v < KW; ++v){
      const float* mp = mrow + (size_t)v*EE;
      const float* wp = wrow + (size_t)v*EE;
      #pragma unroll
      for (int j = 0; j < 4; ++j)
        acc += mp[lane + 32*j] * wp[lane + 32*j];
    }
  }
  #pragma unroll
  for (int off = 16; off > 0; off >>= 1) acc += __shfl_xor(acc, off, 32);
  if (lane == 0){
    raw_ws[gid] = acc;
    raws_out[(((size_t)(b*NSTEP + s)*OO + o)*MM) + (yy+VR)*MSZ + (xx+VR)] = acc;
  }
}

// ---------------- softmax over 4*27*27, write padded sp + next pose ----------------
__global__ void __launch_bounds__(256)
k_softmax(const float* __restrict__ raw_ws, float* __restrict__ sps_out,
          float* __restrict__ pose, int s){
  __shared__ float red[256];
  const int b = blockIdx.x, tid = threadIdx.x;
  const int n = OO*LOC*LOC;                 // 2916
  const float* rb = raw_ws + (size_t)b*n;
  float mx = -1e30f;
  for (int i = tid; i < n; i += 256) mx = fmaxf(mx, rb[i]);
  red[tid] = mx; __syncthreads();
  for (int st = 128; st > 0; st >>= 1){
    if (tid < st) red[tid] = fmaxf(red[tid], red[tid+st]);
    __syncthreads();
  }
  mx = red[0]; __syncthreads();
  float sum = 0.f;
  for (int i = tid; i < n; i += 256) sum += expf(rb[i] - mx);
  red[tid] = sum; __syncthreads();
  for (int st = 128; st > 0; st >>= 1){
    if (tid < st) red[tid] += red[tid+st];
    __syncthreads();
  }
  float inv = 1.0f / red[0];
  for (int i = tid; i < n; i += 256){
    float v = expf(rb[i] - mx) * inv;
    int xx = i % LOC; int t1 = i / LOC;
    int yy = t1 % LOC; int o = t1 / LOC;
    sps_out[(((size_t)(b*NSTEP + s)*OO + o)*MM) + (yy+VR)*MSZ + (xx+VR)] = v;
    pose[((b*OO + o)*MSZ + yy+VR)*MSZ + (xx+VR)] = v;
  }
}

// ---------------- driver ----------------
extern "C" void kernel_launch(void* const* d_in, const int* in_sizes, int n_in,
                              void* d_out, int out_size, void* d_ws, size_t ws_size,
                              hipStream_t stream) {
  const float* images = (const float*)d_in[0];
  const float* wih    = (const float*)d_in[1];
  const float* whh    = (const float*)d_in[2];
  const float* bih    = (const float*)d_in[3];
  const float* bhh    = (const float*)d_in[4];

  float* raws = (float*)d_out;                         // [16,15,4,41,41]
  float* sps  = raws + (size_t)BB*NSTEP*OO*MM;         // [16,15,4,41,41]
  float* maps = sps  + (size_t)BB*NSTEP*OO*MM;         // [16,15,128,41,41]

  char* wp = (char*)d_ws;
  auto alloc = [&](size_t bytes) -> void* {
    void* p = (void*)wp; wp += (bytes + 255) & ~(size_t)255; return p;
  };
  float*  X      = (float*) alloc((size_t)NX*EE*4);
  float*  H      = (float*) alloc((size_t)NX*EE*4);
  float*  C      = (float*) alloc((size_t)NX*EE*4);
  __bf16* Abuf   = (__bf16*)alloc((size_t)NX*256*2);
  __bf16* Wcat   = (__bf16*)alloc((size_t)512*256*2);
  float*  pose   = (float*) alloc((size_t)BB*OO*MM*4);
  float*  wreg   = (float*) alloc((size_t)BB*OO*KK*EE*4);
  float*  wloc   = (float*) alloc((size_t)BB*OO*KK*EE*4);
  float*  raw_ws = (float*) alloc((size_t)BB*OO*LOC*LOC*4);

  // deterministic zero-init each call (graph-capturable memsets)
  hipMemsetAsync(H,    0, (size_t)NX*EE*4, stream);
  hipMemsetAsync(C,    0, (size_t)NX*EE*4, stream);
  hipMemsetAsync(pose, 0, (size_t)BB*OO*MM*4, stream);
  hipMemsetAsync(d_out, 0, (size_t)2*BB*NSTEP*OO*MM*4, stream);  // raw/sp borders

  k_init_pose<<<1, 32, 0, stream>>>(pose);
  k_wcat<<<512, 256, 0, stream>>>(wih, whh, Wcat);

  for (int s = 0; s < NSTEP; ++s){
    k_prep_w<<<(2*BB*OO*KK*EE + 255)/256, 256, 0, stream>>>(images, wreg, wloc, s);
    dim3 rg((MM + 15)/16, BB);
    k_register<<<rg, 256, 0, stream>>>(pose, wreg, X);
    k_pack<<<(int)(((size_t)NX*256 + 255)/256), 256, 0, stream>>>(X, H, Abuf);
    k_gemm_lstm<<<MM, 256, 0, stream>>>(Abuf, Wcat, bih, bhh, H, C, maps, s);
    k_localize<<<(BB*OO*LOC*LOC + 7)/8, 256, 0, stream>>>(H, wloc, raw_ws, raws, s);
    k_softmax<<<BB, 256, 0, stream>>>(raw_ws, sps, pose, s);
  }
}
